// NNCRF_40613210751349
// MI455X (gfx1250) — compile-verified, matrix operand order checked by reading, not run on hardware
//
#include <hip/hip_runtime.h>

// ---------------------------------------------------------------------------
// NNCRF forward for MI455X (gfx1250, wave32, WMMA f32_16x16x32_f16)
// All GEMMs (char-LSTM, syn-LSTM, sa-BiLSTM, GCN, heads) run through
// v_wmma_f32_16x16x32_f16 with weights pre-packed into the native wave32
// B-fragment layout (one 32B contiguous load per lane per fragment).
// ---------------------------------------------------------------------------

#define GS(n) (((n) + 255) / 256)

typedef __attribute__((ext_vector_type(16))) _Float16 v16h;
typedef __attribute__((ext_vector_type(8)))  float    v8f;

union FragA { v16h v; unsigned int u[8]; };
union FragC { v8f  v; float f[8]; };
union PackH { _Float16 h[2]; unsigned int u; };

static __device__ __forceinline__ float sigm(float x){ return 1.0f / (1.0f + __expf(-x)); }

// A fragment (16x32 f16, M x K) from an LDS tile stored row-major [16][lda]
static __device__ __forceinline__ v16h load_a_frag(const _Float16* Al, int lda, int lane, int kc){
  int m = lane & 15, half = lane >> 4;
  FragA a;
#pragma unroll
  for (int v = 0; v < 8; ++v){
    int k = kc * 32 + ((v < 4) ? (2 * v) : (2 * v + 8)) + 8 * half;
    a.u[v] = *reinterpret_cast<const unsigned int*>(&Al[m * lda + k]);
  }
  return a.v;
}

// B fragment (32x16 f16, K x N) from fragment-native packed weights:
// tile (nt,kc) is 512 contiguous halves, lane's 16 halves are contiguous.
static __device__ __forceinline__ v16h load_b_frag(const _Float16* Bp, int KC, int nt, int kc, int lane){
  return *reinterpret_cast<const v16h*>(Bp + (((size_t)nt * KC + kc) * 32 + lane) * 16);
}

static __device__ __forceinline__ v8f wmma_f16(v16h a, v16h b, v8f c){
  return __builtin_amdgcn_wmma_f32_16x16x32_f16(false, a, false, b, (short)0, c, false, false);
}

// ---------------------------------------------------------------------------
// Utility kernels
// ---------------------------------------------------------------------------
__global__ void k_zero_f(float* p, int n){
  int i = blockIdx.x * blockDim.x + threadIdx.x; if (i < n) p[i] = 0.f;
}
__global__ void k_zero_h(_Float16* p, int n){
  int i = blockIdx.x * blockDim.x + threadIdx.x; if (i < n) p[i] = (_Float16)0.f;
}
__global__ void k_fill_f(float* p, float v, int n){
  int i = blockIdx.x * blockDim.x + threadIdx.x; if (i < n) p[i] = v;
}

// Pack W[n][inOff+k] (row-major f32 [outCnt x ldW]) into fragment-native f16
// layout at tile offset (k0, n0).  Wp must be pre-zeroed (covers pads).
__global__ void k_pack_w(const float* __restrict__ W, int ldW, int inOff,
                         int k0, int n0, int inCnt, int outCnt,
                         _Float16* __restrict__ Wp, int KC){
  int idx = blockIdx.x * blockDim.x + threadIdx.x;
  if (idx >= inCnt * outCnt) return;
  int kk = idx % inCnt, nn = idx / inCnt;
  float val = W[(size_t)nn * ldW + inOff + kk];
  int k = k0 + kk, n = n0 + nn;
  int kc = k >> 5, kr = k & 31, nt = n >> 4, nr = n & 15;
  int half = (kr >> 3) & 1;
  int v = ((kr & 16) >> 2) + ((kr & 7) >> 1);
  int lane = nr + 16 * half;
  size_t o = (((size_t)nt * KC + kc) * 32 + lane) * 16 + v * 2 + (kr & 1);
  Wp[o] = (_Float16)val;
}

// dst[g*ps + r] = a[g*gs + r] (+ b[g*gs + r])   for g in [0,ng)
__global__ void k_bias_pad(float* __restrict__ dst, const float* __restrict__ a,
                           const float* __restrict__ b, int gs, int ps, int ng){
  int idx = blockIdx.x * blockDim.x + threadIdx.x;
  if (idx >= gs * ng) return;
  int g = idx / gs, r = idx % gs;
  float v = a[idx];
  if (b) v += b[idx];
  dst[g * ps + r] = v;
}

// word embedding gather into word_features[:, 0:300]
__global__ void k_word_embed(const int* __restrict__ wi, const float* __restrict__ emb,
                             float* __restrict__ wf){
  int idx = blockIdx.x * blockDim.x + threadIdx.x;
  if (idx >= 4096 * 300) return;
  int n = idx / 300, c = idx % 300;
  wf[(size_t)n * 400 + c] = emb[(size_t)wi[n] * 300 + c];
}

// ---------------------------------------------------------------------------
// Char BiLSTM (packed semantics): forward scan, capture h at len-1; backward
// is one step from zero state on x[len-1].  16 rows/block, 4 waves, WMMA.
// A tile layout: k 0..49 = x (pad to 64), k 64..113 = h (pad to 128).
// Gate columns padded: n = g*64 + j, j<50 valid, 4 gates (i,f,g,o).
// ---------------------------------------------------------------------------
__global__ __launch_bounds__(128) void k_char_lstm(
    const int* __restrict__ ci, const int* __restrict__ clens,
    const float* __restrict__ cemb,
    const _Float16* __restrict__ WpF, const float* __restrict__ bF,
    const _Float16* __restrict__ WpB, const float* __restrict__ bB,
    float* __restrict__ wf){
  const int LDA = 136, KC = 4;
  __shared__ _Float16 A[16 * LDA];
  __shared__ int lens[16];
  int tid = threadIdx.x, lane = tid & 31, wave = tid >> 5;
  int rowbase = blockIdx.x * 16;
  for (int i = tid; i < 16 * LDA; i += 128) A[i] = (_Float16)0.f;
  if (tid < 16) lens[tid] = clens[rowbase + tid];
  __syncthreads();
  int jt = wave;                       // 0..3 : 16-col slice of each gate
  int m = lane & 15, half = lane >> 4;
  int j = jt * 16 + m;                 // 0..63 within gate
  FragC cfr;
#pragma unroll
  for (int v = 0; v < 8; ++v) cfr.f[v] = 0.f;
  float bg[4];
#pragma unroll
  for (int g = 0; g < 4; ++g) bg[g] = bF[g * 64 + j];

  for (int t = 0; t < 16; ++t){
    for (int i = tid; i < 16 * 50; i += 128){
      int mm = i / 50, cc = i % 50;
      int cidx = ci[(rowbase + mm) * 16 + t];
      A[mm * LDA + cc] = (_Float16)cemb[cidx * 50 + cc];
    }
    __syncthreads();
    FragC acc[4];
#pragma unroll
    for (int g = 0; g < 4; ++g){
      v8f z = {0.f,0.f,0.f,0.f,0.f,0.f,0.f,0.f};
#pragma unroll
      for (int kc = 0; kc < KC; ++kc){
        v16h a = load_a_frag(A, LDA, lane, kc);
        v16h b = load_b_frag(WpF, KC, g * 4 + jt, kc, lane);
        z = wmma_f16(a, b, z);
      }
      acc[g].v = z;
    }
    __syncthreads();                   // everyone done reading A
#pragma unroll
    for (int v = 0; v < 8; ++v){
      int mm = v + 8 * half;
      float iv = sigm(acc[0].f[v] + bg[0]);
      float fv = sigm(acc[1].f[v] + bg[1]);
      float gv = tanhf(acc[2].f[v] + bg[2]);
      float ov = sigm(acc[3].f[v] + bg[3]);
      float c = fv * cfr.f[v] + iv * gv;
      cfr.f[v] = c;
      float h = ov * tanhf(c);
      A[mm * LDA + 64 + j] = (_Float16)h;                       // h for next step
      if (j < 50 && lens[mm] - 1 == t)
        wf[(size_t)(rowbase + mm) * 400 + 300 + j] = h;         // h_last_f
    }
    __syncthreads();
  }
  // backward: single step from zero state on x[len-1]
  for (int i = tid; i < 16 * 64; i += 128){
    int mm = i / 64, cc = i % 64;
    A[mm * LDA + 64 + cc] = (_Float16)0.f;
  }
  __syncthreads();
  for (int i = tid; i < 16 * 50; i += 128){
    int mm = i / 50, cc = i % 50;
    int tl = lens[mm] - 1;
    int cidx = ci[(rowbase + mm) * 16 + tl];
    A[mm * LDA + cc] = (_Float16)cemb[cidx * 50 + cc];
  }
  __syncthreads();
#pragma unroll
  for (int g = 0; g < 4; ++g) bg[g] = bB[g * 64 + j];
  FragC accb[4];
#pragma unroll
  for (int g = 0; g < 4; ++g){
    v8f z = {0.f,0.f,0.f,0.f,0.f,0.f,0.f,0.f};
#pragma unroll
    for (int kc = 0; kc < KC; ++kc){
      v16h a = load_a_frag(A, LDA, lane, kc);
      v16h b = load_b_frag(WpB, KC, g * 4 + jt, kc, lane);
      z = wmma_f16(a, b, z);
    }
    accb[g].v = z;
  }
#pragma unroll
  for (int v = 0; v < 8; ++v){
    int mm = v + 8 * half;
    float iv = sigm(accb[0].f[v] + bg[0]);
    float gv = tanhf(accb[2].f[v] + bg[2]);
    float ov = sigm(accb[3].f[v] + bg[3]);
    float h = ov * tanhf(iv * gv);     // c_prev = 0 => f-term vanishes
    if (j < 50)
      wf[(size_t)(rowbase + mm) * 400 + 350 + j] = h;           // h_last_b
  }
}

// ---------------------------------------------------------------------------
// Syn-LSTM (MyLSTM), persistent: 16 batch rows/block, 8 waves, 128 steps.
// A tile K layout: [h 0..399][wf 400..799][gcn 800..999][gcn(m) 1000..1199][pad]
// Fused weight N layout: [i][o][f][u][ii][uu], each 400 wide (25 tiles).
// ---------------------------------------------------------------------------
__global__ __launch_bounds__(256) void k_syn_lstm(
    const float* __restrict__ wf, const float* __restrict__ gcn,
    const _Float16* __restrict__ Wp, const float* __restrict__ bb,
    float* __restrict__ hs){
  const int LDA = 1224, KC = 38;
  __shared__ _Float16 A[16 * LDA];
  int tid = threadIdx.x, lane = tid & 31, wave = tid >> 5;
  int rowbase = blockIdx.x * 16;
  int m = lane & 15, half = lane >> 4;
  for (int i = tid; i < 16 * LDA; i += 256) A[i] = (_Float16)0.f;
  __syncthreads();
  FragC cfr[4];
  float hreg[4][8];
#pragma unroll
  for (int ji = 0; ji < 4; ++ji)
#pragma unroll
    for (int v = 0; v < 8; ++v) cfr[ji].f[v] = 0.f;

  for (int t = 0; t < 128; ++t){
    for (int i = tid; i < 16 * 800; i += 256){
      int mm = i / 800, c = i % 800;
      size_t node = (size_t)(rowbase + mm) * 128 + t;
      float val = (c < 400) ? wf[node * 400 + c]
                            : gcn[node * 200 + ((c < 600) ? c - 400 : c - 600)];
      A[mm * LDA + 400 + c] = (_Float16)val;
    }
    __syncthreads();
#pragma unroll
    for (int ji = 0; ji < 4; ++ji){
      int jt = wave + ji * 8;
      if (jt >= 25) continue;          // wave-uniform
      int j = jt * 16 + m;
      FragC acc[6];
#pragma unroll
      for (int g = 0; g < 6; ++g){ v8f z = {0.f,0.f,0.f,0.f,0.f,0.f,0.f,0.f}; acc[g].v = z; }
      for (int kc = 0; kc < KC; ++kc){
        v16h a = load_a_frag(A, LDA, lane, kc);
        if (kc + 1 < KC)
          __builtin_prefetch(Wp + (((size_t)jt * KC + kc + 1) * 32 + lane) * 16, 0, 3);
#pragma unroll
        for (int g = 0; g < 6; ++g){
          v16h b = load_b_frag(Wp, KC, g * 25 + jt, kc, lane);
          acc[g].v = wmma_f16(a, b, acc[g].v);
        }
      }
      float bg[6];
#pragma unroll
      for (int g = 0; g < 6; ++g) bg[g] = bb[g * 400 + j];
#pragma unroll
      for (int v = 0; v < 8; ++v){
        float i_ = sigm(acc[0].f[v] + bg[0]);
        float o_ = sigm(acc[1].f[v] + bg[1]);
        float f_ = sigm(acc[2].f[v] + bg[2]);
        float u_ = tanhf(acc[3].f[v] + bg[3]);
        float i2 = sigm(acc[4].f[v] + bg[4]);
        float u2 = tanhf(acc[5].f[v] + bg[5]);
        float c = i_ * u_ + i2 * u2 + f_ * cfr[ji].f[v];
        cfr[ji].f[v] = c;
        hreg[ji][v] = o_ * tanhf(c);
      }
    }
    __syncthreads();                   // all reads of A done before h update
#pragma unroll
    for (int ji = 0; ji < 4; ++ji){
      int jt = wave + ji * 8;
      if (jt >= 25) continue;
      int j = jt * 16 + m;
#pragma unroll
      for (int v = 0; v < 8; ++v){
        int mm = v + 8 * half;
        float h = hreg[ji][v];
        A[mm * LDA + j] = (_Float16)h;
        hs[((size_t)(rowbase + mm) * 128 + t) * 400 + j] = h;
      }
    }
    __syncthreads();
  }
}

// ---------------------------------------------------------------------------
// Standard LSTM direction (sa-lstm layers): 16 rows/block, dir = blockIdx.y.
// A K layout: [x 0..399][h 400..599][pad].  Gates padded to 208 (13 tiles).
// ---------------------------------------------------------------------------
__global__ __launch_bounds__(256) void k_sa_lstm(
    const float* __restrict__ xin,
    const _Float16* __restrict__ WpF, const float* __restrict__ bF,
    const _Float16* __restrict__ WpB, const float* __restrict__ bB,
    float* __restrict__ xout){
  const int LDA = 616, KC = 19;
  __shared__ _Float16 A[16 * LDA];
  int tid = threadIdx.x, lane = tid & 31, wave = tid >> 5;
  int rowbase = blockIdx.x * 16;
  int dir = blockIdx.y;
  const _Float16* Wp = dir ? WpB : WpF;
  const float* bb = dir ? bB : bF;
  int m = lane & 15, half = lane >> 4;
  for (int i = tid; i < 16 * LDA; i += 256) A[i] = (_Float16)0.f;
  __syncthreads();
  FragC cfr[2];
  float hreg[2][8];
#pragma unroll
  for (int ji = 0; ji < 2; ++ji)
#pragma unroll
    for (int v = 0; v < 8; ++v) cfr[ji].f[v] = 0.f;

  for (int t = 0; t < 128; ++t){
    int tt = dir ? (127 - t) : t;
    for (int i = tid; i < 16 * 400; i += 256){
      int mm = i / 400, c = i % 400;
      A[mm * LDA + c] = (_Float16)xin[((size_t)(rowbase + mm) * 128 + tt) * 400 + c];
    }
    __syncthreads();
#pragma unroll
    for (int ji = 0; ji < 2; ++ji){
      int jt = wave + ji * 8;
      if (jt >= 13) continue;
      int j = jt * 16 + m;
      FragC acc[4];
#pragma unroll
      for (int g = 0; g < 4; ++g){ v8f z = {0.f,0.f,0.f,0.f,0.f,0.f,0.f,0.f}; acc[g].v = z; }
      for (int kc = 0; kc < KC; ++kc){
        v16h a = load_a_frag(A, LDA, lane, kc);
#pragma unroll
        for (int g = 0; g < 4; ++g){
          v16h b = load_b_frag(Wp, KC, g * 13 + jt, kc, lane);
          acc[g].v = wmma_f16(a, b, acc[g].v);
        }
      }
      float bg[4];
#pragma unroll
      for (int g = 0; g < 4; ++g) bg[g] = bb[g * 208 + j];
#pragma unroll
      for (int v = 0; v < 8; ++v){
        float iv = sigm(acc[0].f[v] + bg[0]);
        float fv = sigm(acc[1].f[v] + bg[1]);
        float gv = tanhf(acc[2].f[v] + bg[2]);
        float ov = sigm(acc[3].f[v] + bg[3]);
        float c = fv * cfr[ji].f[v] + iv * gv;
        cfr[ji].f[v] = c;
        hreg[ji][v] = ov * tanhf(c);
      }
    }
    __syncthreads();
#pragma unroll
    for (int ji = 0; ji < 2; ++ji){
      int jt = wave + ji * 8;
      if (jt >= 13) continue;
      int j = jt * 16 + m;
      if (j < 200){
#pragma unroll
        for (int v = 0; v < 8; ++v){
          int mm = v + 8 * half;
          float h = hreg[ji][v];
          A[mm * LDA + 400 + j] = (_Float16)h;
          xout[((size_t)(rowbase + mm) * 128 + tt) * 400 + dir * 200 + j] = h;
        }
      }
    }
    __syncthreads();
  }
}

// ---------------------------------------------------------------------------
// Generic WMMA GEMM:  C[M x Nout] = A[M x K] * Bp + bias, optional ReLU.
// grid.x = M/16, grid.y covers NT/8; 8 waves/block, one n-tile per wave.
// ---------------------------------------------------------------------------
__global__ __launch_bounds__(256) void k_gemm(
    const float* __restrict__ Aglob, int lda,
    const _Float16* __restrict__ Bp, const float* __restrict__ bias,
    float* __restrict__ C, int ldc,
    int K, int Kpad, int NT, int Nout, int relu){
  __shared__ _Float16 A[16 * 1232];
  const int LDA = Kpad + 16;
  int tid = threadIdx.x, lane = tid & 31, wave = tid >> 5;
  int row0 = blockIdx.x * 16;
  int KC = Kpad >> 5;
  for (int i = tid; i < 16 * (Kpad >> 1); i += 256){
    int mm = i / (Kpad >> 1);
    int k2 = (i % (Kpad >> 1)) * 2;
    PackH pk;
    pk.h[0] = (_Float16)((k2     < K) ? Aglob[(size_t)(row0 + mm) * lda + k2]     : 0.f);
    pk.h[1] = (_Float16)((k2 + 1 < K) ? Aglob[(size_t)(row0 + mm) * lda + k2 + 1] : 0.f);
    *reinterpret_cast<unsigned int*>(&A[mm * LDA + k2]) = pk.u;
  }
  __syncthreads();
  int m = lane & 15, half = lane >> 4;
  for (int nt = blockIdx.y * 8 + wave; nt < NT; nt += gridDim.y * 8){
    v8f z = {0.f,0.f,0.f,0.f,0.f,0.f,0.f,0.f};
    for (int kc = 0; kc < KC; ++kc){
      v16h a = load_a_frag(A, LDA, lane, kc);
      if (kc + 1 < KC)
        __builtin_prefetch(Bp + (((size_t)nt * KC + kc + 1) * 32 + lane) * 16, 0, 3);
      v16h b = load_b_frag(Bp, KC, nt, kc, lane);
      z = wmma_f16(a, b, z);
    }
    FragC acc; acc.v = z;
    int n = nt * 16 + m;
    if (n < Nout){
      float bv = bias ? bias[n] : 0.f;
#pragma unroll
      for (int v = 0; v < 8; ++v){
        int mm = v + 8 * half;
        float val = acc.f[v] + bv;
        if (relu) val = fmaxf(val, 0.f);
        C[(size_t)(row0 + mm) * ldc + n] = val;
      }
    }
  }
}

// ---------------------------------------------------------------------------
// GCN / scatter kernels
// ---------------------------------------------------------------------------
__global__ void k_scatter_dep(const int* __restrict__ col, const int* __restrict__ lab,
                              const float* __restrict__ demb, float* __restrict__ dexp){
  int idx = blockIdx.x * blockDim.x + threadIdx.x;
  if (idx >= 8192 * 64) return;
  int e = idx >> 6, c = idx & 63;
  atomicAdd(&dexp[(size_t)col[e] * 64 + c], demb[lab[e] * 64 + c]);
}
__global__ void k_deg_edges(const int* __restrict__ col, float* __restrict__ deg){
  int e = blockIdx.x * blockDim.x + threadIdx.x;
  if (e < 8192) atomicAdd(&deg[col[e]], 1.f);
}
__global__ void k_rsqrt(float* p, int n){
  int i = blockIdx.x * blockDim.x + threadIdx.x; if (i < n) p[i] = rsqrtf(p[i]);
}
__global__ void k_concat_xg(const float* __restrict__ wf, const float* __restrict__ dexp,
                            float* __restrict__ xg){
  int idx = blockIdx.x * blockDim.x + threadIdx.x;
  if (idx >= 4096 * 464) return;
  int n = idx / 464, c = idx % 464;
  xg[idx] = (c < 400) ? wf[(size_t)n * 400 + c] : dexp[(size_t)n * 64 + (c - 400)];
}
__global__ void k_gcn_self(const float* __restrict__ xw, const float* __restrict__ dis,
                           float* __restrict__ gcn){
  int idx = blockIdx.x * blockDim.x + threadIdx.x;
  if (idx >= 4096 * 200) return;
  int n = idx / 200;
  gcn[idx] = dis[n] * dis[n] * xw[idx];
}
__global__ void k_gcn_edges(const int* __restrict__ row, const int* __restrict__ col,
                            const float* __restrict__ dis, const float* __restrict__ xw,
                            float* __restrict__ gcn){
  int idx = blockIdx.x * blockDim.x + threadIdx.x;
  if (idx >= 8192 * 200) return;
  int e = idx / 200, c = idx % 200;
  int r = row[e], cl = col[e];
  atomicAdd(&gcn[(size_t)cl * 200 + c], dis[r] * dis[cl] * xw[(size_t)r * 200 + c]);
}
__global__ void k_bias_relu(float* __restrict__ g, const float* __restrict__ b, int n, int D){
  int i = blockIdx.x * blockDim.x + threadIdx.x;
  if (i < n) g[i] = fmaxf(g[i] + b[i % D], 0.f);
}

// ---------------------------------------------------------------------------
// Heads
// ---------------------------------------------------------------------------
__global__ void k_pool(const float* __restrict__ x, float* __restrict__ pooled){
  int idx = blockIdx.x * blockDim.x + threadIdx.x;
  if (idx >= 32 * 400) return;
  int b = idx / 400, c = idx % 400;
  float s = 0.f, mx = -3.402823466e38f, lastv = 0.f;
  for (int t = 0; t < 128; ++t){
    float v = x[((size_t)b * 128 + t) * 400 + c];
    s += v; mx = fmaxf(mx, v); lastv = v;
  }
  pooled[b * 1200 + c]       = s * (1.f / 128.f);
  pooled[b * 1200 + 400 + c] = mx;
  pooled[b * 1200 + 800 + c] = lastv;
}
__global__ __launch_bounds__(256) void k_ln_relu(float* __restrict__ x,
                                                 const float* __restrict__ g,
                                                 const float* __restrict__ be, int R, int D){
  int wave = (blockIdx.x * blockDim.x + threadIdx.x) >> 5;
  int lane = threadIdx.x & 31;
  if (wave >= R) return;
  float* row = x + (size_t)wave * D;
  float s = 0.f, s2 = 0.f;
  for (int i = lane; i < D; i += 32){ float v = row[i]; s += v; s2 += v * v; }
#pragma unroll
  for (int o = 16; o > 0; o >>= 1){ s += __shfl_xor(s, o, 32); s2 += __shfl_xor(s2, o, 32); }
  float mean = s / D;
  float var  = s2 / D - mean * mean;
  float inv  = rsqrtf(var + 1e-5f);
  for (int i = lane; i < D; i += 32){
    float v = (row[i] - mean) * inv * g[i] + be[i];
    row[i] = fmaxf(v, 0.f);
  }
}
__global__ void k_final_fc(const float* __restrict__ h2, const float* __restrict__ W,
                           const float* __restrict__ b, float* __restrict__ out){
  int idx = blockIdx.x * blockDim.x + threadIdx.x;
  if (idx >= 64) return;
  int bt = idx >> 1, o = idx & 1;
  float s = b[o];
  for (int i = 0; i < 64; ++i) s += h2[bt * 64 + i] * W[o * 64 + i];
  out[bt * 2 + o] = s;
}

// ---------------------------------------------------------------------------
extern "C" void kernel_launch(void* const* d_in, const int* in_sizes, int n_in,
                              void* d_out, int out_size, void* d_ws, size_t ws_size,
                              hipStream_t stream){
  (void)in_sizes; (void)n_in; (void)out_size; (void)ws_size;
  const int* word_inputs = (const int*)d_in[0];
  const int* char_inputs = (const int*)d_in[1];
  const int* char_lens   = (const int*)d_in[2];
  const int* edge_row    = (const int*)d_in[3];
  const int* edge_col    = ((const int*)d_in[3]) + 8192;
  const int* dep_labels  = (const int*)d_in[4];
  auto F = [&](int i){ return (const float*)d_in[i]; };
  // params (setup_inputs dict order, offset by 5 inputs)
  const float *word_emb=F(5), *char_emb=F(6);
  const float *cWihF=F(7), *cWhhF=F(8), *cbihF=F(9), *cbhhF=F(10);
  const float *cWihB=F(11), *cWhhB=F(12), *cbihB=F(13), *cbhhB=F(14);
  const float *dep_emb=F(15), *gcn_W=F(16), *gcn_b=F(17);
  const float *synW[6] = {F(18), F(20), F(22), F(24), F(26), F(28)};
  const float *synB[6] = {F(19), F(21), F(23), F(25), F(27), F(29)};
  const float *saWih[4]={F(30),F(34),F(38),F(42)};
  const float *saWhh[4]={F(31),F(35),F(39),F(43)};
  const float *saBih[4]={F(32),F(36),F(40),F(44)};
  const float *saBhh[4]={F(33),F(37),F(41),F(45)};
  const float *nerW1=F(46),*nerB1=F(47),*nerG=F(48),*nerBe=F(49),*nerW2=F(50),*nerB2=F(51);
  const float *sahW1=F(52),*sahB1=F(53),*sahG=F(54),*sahBe=F(55),*sahW2=F(56),*sahB2=F(57);

  // workspace carve (64B aligned)
  char* base = (char*)d_ws; size_t off = 0;
  auto alloc = [&](size_t bytes)->char*{
    off = (off + 63) & ~(size_t)63; char* p = base + off; off += bytes; return p;
  };
  // f16 packed weights (contiguous block, zeroed once)
  _Float16* WpCharF = (_Float16*)alloc(32768 * 2);
  _Float16* WpCharB = (_Float16*)alloc(32768 * 2);
  _Float16* WpSyn   = (_Float16*)alloc((size_t)2918400 * 2);
  _Float16* WpSa[4];
  for (int i = 0; i < 4; ++i) WpSa[i] = (_Float16*)alloc(505856 * 2);
  _Float16* WpGcn  = (_Float16*)alloc(99840 * 2);
  _Float16* WpNer1 = (_Float16*)alloc(86528 * 2);
  _Float16* WpNer2 = (_Float16*)alloc(7168 * 2);
  _Float16* WpSah  = (_Float16*)alloc(77824 * 2);
  const int totHalves = 32768*2 + 2918400 + 505856*4 + 99840 + 86528 + 7168 + 77824;
  // padded biases (contiguous)
  float* bCharF = (float*)alloc(256 * 4);
  float* bCharB = (float*)alloc(256 * 4);
  float* bSyn   = (float*)alloc(2400 * 4);
  float* bSa[4];
  for (int i = 0; i < 4; ++i) bSa[i] = (float*)alloc(832 * 4);
  const int totBias = 256 + 256 + 2400 + 832 * 4;
  // activations
  float* wf     = (float*)alloc((size_t)4096 * 400 * 4);
  float* dexp   = (float*)alloc((size_t)4096 * 64 * 4);
  float* xg     = (float*)alloc((size_t)4096 * 464 * 4);
  float* xw     = (float*)alloc((size_t)4096 * 200 * 4);
  float* deg    = (float*)alloc(4096 * 4);
  float* gcn    = (float*)alloc((size_t)4096 * 200 * 4);
  float* hs     = (float*)alloc((size_t)4096 * 400 * 4);
  float* nerH1  = (float*)alloc((size_t)4096 * 200 * 4);
  float* saXa   = (float*)alloc((size_t)4096 * 400 * 4);
  float* saXb   = (float*)alloc((size_t)4096 * 400 * 4);
  float* pooled = (float*)alloc(32 * 1200 * 4);
  float* sahH1  = (float*)alloc(32 * 64 * 4);
  float* em_ner = (float*)d_out;            // [32,128,20]
  float* em_sa  = (float*)d_out + 81920;    // [32,2]

  auto pack = [&](const float* W, int ldW, int inOff, int k0, int n0,
                  int inCnt, int outCnt, _Float16* Wp, int KC){
    int tot = inCnt * outCnt;
    k_pack_w<<<GS(tot), 256, 0, stream>>>(W, ldW, inOff, k0, n0, inCnt, outCnt, Wp, KC);
  };

  // ---- weight / bias packing -------------------------------------------------
  k_zero_h<<<GS(totHalves), 256, 0, stream>>>(WpCharF, totHalves);
  k_zero_f<<<GS(totBias), 256, 0, stream>>>(bCharF, totBias);
  for (int g = 0; g < 4; ++g){
    pack(cWihF + g*50*50, 50, 0, 0,  g*64, 50, 50, WpCharF, 4);
    pack(cWhhF + g*50*50, 50, 0, 64, g*64, 50, 50, WpCharF, 4);
    pack(cWihB + g*50*50, 50, 0, 0,  g*64, 50, 50, WpCharB, 4);
    pack(cWhhB + g*50*50, 50, 0, 64, g*64, 50, 50, WpCharB, 4);
  }
  // syn fused weight: k layout [h 0..399][x 400..999][m 1000..1199]
  pack(synW[0], 1000, 0, 0,    0,    1000, 400, WpSyn, 38);  // W1 (i)
  pack(synW[1], 1200, 0, 0,    400,  1200, 400, WpSyn, 38);  // W2 (o)
  pack(synW[2], 1200, 0, 0,    800,  1200, 400, WpSyn, 38);  // W3 (f)
  pack(synW[3], 1000, 0, 0,    1200, 1000, 400, WpSyn, 38);  // W4 (u)
  pack(synW[4], 600,  0, 0,    1600, 400,  400, WpSyn, 38);  // W11 h part
  pack(synW[4], 600,  400, 1000, 1600, 200, 400, WpSyn, 38); // W11 m part
  pack(synW[5], 600,  0, 0,    2000, 400,  400, WpSyn, 38);  // W44 h part
  pack(synW[5], 600,  400, 1000, 2000, 200, 400, WpSyn, 38); // W44 m part
  for (int i = 0; i < 4; ++i)
    for (int g = 0; g < 4; ++g){
      pack(saWih[i] + (size_t)g*200*400, 400, 0, 0,   g*208, 400, 200, WpSa[i], 19);
      pack(saWhh[i] + (size_t)g*200*200, 200, 0, 400, g*208, 200, 200, WpSa[i], 19);
    }
  pack(gcn_W, 464, 0, 0, 0, 464, 200, WpGcn, 15);
  pack(nerW1, 400, 0, 0, 0, 400, 200, WpNer1, 13);
  pack(nerW2, 200, 0, 0, 0, 200, 20,  WpNer2, 7);
  pack(sahW1, 1200, 0, 0, 0, 1200, 64, WpSah, 38);
  k_bias_pad<<<GS(200), 256, 0, stream>>>(bCharF, cbihF, cbhhF, 50, 64, 4);
  k_bias_pad<<<GS(200), 256, 0, stream>>>(bCharB, cbihB, cbhhB, 50, 64, 4);
  for (int g = 0; g < 6; ++g)
    k_bias_pad<<<GS(400), 256, 0, stream>>>(bSyn + g*400, synB[g], nullptr, 400, 400, 1);
  for (int i = 0; i < 4; ++i)
    k_bias_pad<<<GS(800), 256, 0, stream>>>(bSa[i], saBih[i], saBhh[i], 200, 208, 4);

  // ---- forward ---------------------------------------------------------------
  k_word_embed<<<GS(4096*300), 256, 0, stream>>>(word_inputs, word_emb, wf);
  k_char_lstm<<<256, 128, 0, stream>>>(char_inputs, char_lens, char_emb,
                                       WpCharF, bCharF, WpCharB, bCharB, wf);
  // GCN input
  k_zero_f<<<GS(4096*64), 256, 0, stream>>>(dexp, 4096*64);
  k_scatter_dep<<<GS(8192*64), 256, 0, stream>>>(edge_col, dep_labels, dep_emb, dexp);
  k_fill_f<<<GS(4096), 256, 0, stream>>>(deg, 1.0f, 4096);
  k_deg_edges<<<GS(8192), 256, 0, stream>>>(edge_col, deg);
  k_rsqrt<<<GS(4096), 256, 0, stream>>>(deg, 4096);
  k_concat_xg<<<GS(4096*464), 256, 0, stream>>>(wf, dexp, xg);
  k_gemm<<<dim3(256, 2), 256, 0, stream>>>(xg, 464, WpGcn, nullptr, xw, 200,
                                           464, 480, 13, 200, 0);
  k_gcn_self<<<GS(4096*200), 256, 0, stream>>>(xw, deg, gcn);
  k_gcn_edges<<<GS(8192*200), 256, 0, stream>>>(edge_row, edge_col, deg, xw, gcn);
  k_bias_relu<<<GS(4096*200), 256, 0, stream>>>(gcn, gcn_b, 4096*200, 200);
  // syn-LSTM -> hs
  k_syn_lstm<<<2, 256, 0, stream>>>(wf, gcn, WpSyn, bSyn, hs);
  // sa-BiLSTM (2 layers)
  k_sa_lstm<<<dim3(2, 2), 256, 0, stream>>>(wf,  WpSa[0], bSa[0], WpSa[1], bSa[1], saXa);
  k_sa_lstm<<<dim3(2, 2), 256, 0, stream>>>(saXa, WpSa[2], bSa[2], WpSa[3], bSa[3], saXb);
  // NER head
  k_gemm<<<dim3(256, 2), 256, 0, stream>>>(hs, 400, WpNer1, nerB1, nerH1, 200,
                                           400, 416, 13, 200, 0);
  k_ln_relu<<<GS(4096*32), 256, 0, stream>>>(nerH1, nerG, nerBe, 4096, 200);
  k_gemm<<<dim3(256, 1), 256, 0, stream>>>(nerH1, 200, WpNer2, nerB2, em_ner, 20,
                                           200, 224, 2, 20, 0);
  // SA head
  k_pool<<<GS(32*400), 256, 0, stream>>>(saXb, pooled);
  k_gemm<<<dim3(2, 1), 256, 0, stream>>>(pooled, 1200, WpSah, sahB1, sahH1, 64,
                                         1200, 1216, 4, 64, 0);
  k_ln_relu<<<GS(32*32), 256, 0, stream>>>(sahH1, sahG, sahBe, 32, 64);
  k_final_fc<<<1, 64, 0, stream>>>(sahH1, sahW2, sahB2, em_sa);
}
